// CrossAttention_29454885716439
// MI455X (gfx1250) — matrix operand order repaired and checked by reading, not compiled, not run on hardware
//
#include <hip/hip_runtime.h>
#include <hip/hip_bf16.h>
#include <math.h>

typedef __attribute__((ext_vector_type(16))) __bf16 v16bf;
typedef __attribute__((ext_vector_type(8)))  float  v8f;
typedef __attribute__((ext_vector_type(4)))  int    v4i;
typedef unsigned int   u32;
typedef unsigned short u16;

#define BATCH 2
#define NSEQ  2048      // Nq == Nk
#define QD    1024
#define KVD   768
#define ED    1024
#define NH    16
#define HD    64

// SCALE * log2(e) so softmax exp becomes a single v_exp_f32 (exp2)
#define SCALE_LOG2E 0.18033688011112042f

// raw v_exp_f32: argument is provably in [-0.19, 0.19], no denorm guard needed
__device__ __forceinline__ float fast_exp2(float x) {
#if __has_builtin(__builtin_amdgcn_exp2f)
  return __builtin_amdgcn_exp2f(x);
#else
  return exp2f(x);
#endif
}

__device__ __forceinline__ u16 f2b(float f) {          // strict RNE (cold paths)
  u32 u = __float_as_uint(f);
  u32 r = u + 0x7FFFu + ((u >> 16) & 1u);
  return (u16)(r >> 16);
}
__device__ __forceinline__ float b2f(u16 h) {
  return __uint_as_float(((u32)h) << 16);
}
// cheap round-half-up pack of two f32 -> packed bf16x2 (hot loops)
__device__ __forceinline__ u32 pack_bf16(float lo, float hi) {
  u32 ul = __float_as_uint(lo) + 0x8000u;
  u32 uh = __float_as_uint(hi) + 0x8000u;
  return (ul >> 16) | (uh & 0xFFFF0000u);
}

// ---- async global->LDS staging (ASYNCcnt path), with safe fallback --------
#if __has_builtin(__builtin_amdgcn_global_load_async_to_lds_b128) && \
    __has_builtin(__builtin_amdgcn_s_wait_asynccnt)
#define HAVE_ASYNC_LDS 1
__device__ __forceinline__ void async_b128(const u16* g, u16* l) {
  __builtin_amdgcn_global_load_async_to_lds_b128(
      (v4i __attribute__((address_space(1)))*)g,
      (v4i __attribute__((address_space(3)))*)l, 0, 0);
}
__device__ __forceinline__ void async_wait0() {
  __builtin_amdgcn_s_wait_asynccnt(0);
}
#else
#define HAVE_ASYNC_LDS 0
__device__ __forceinline__ void async_b128(const u16* g, u16* l) {
  *(uint4*)l = *(const uint4*)g;
}
__device__ __forceinline__ void async_wait0() {}
#endif

union BF16Frag { v16bf v; uint4 q[2]; u32 w[8]; };

__device__ __forceinline__ v8f wmma_bf16(v16bf a, v16bf b, v8f c) {
  return __builtin_amdgcn_wmma_f32_16x16x32_bf16(false, a, false, b, (short)0, c,
                                                 false, false);
}

// A-fragment (16x32 bf16, M x K): lane li = row; halves = K at lg*8..+7 and lg*8+16..+23
__device__ __forceinline__ v16bf load_a_frag(const u16* rowp, int lg) {
  BF16Frag f;
  f.q[0] = *(const uint4*)(rowp + lg * 8);
  f.q[1] = *(const uint4*)(rowp + lg * 8 + 16);
  return f.v;
}
// B-fragment (32x16 bf16, K x N): lane li = column; halves = K at lg*8..+7 and lg*8+16..+23
__device__ __forceinline__ v16bf load_b_frag(const u16* rowp, int lg) {
  BF16Frag f;
  f.q[0] = *(const uint4*)(rowp + lg * 8);
  f.q[1] = *(const uint4*)(rowp + lg * 8 + 16);
  return f.v;
}

// ---------------------------------------------------------------------------
// 1) weight transpose + f32->bf16:  WT[n*K + k] = bf16(W[k*N + n]),  N = 1024
// ---------------------------------------------------------------------------
__global__ void __launch_bounds__(256)
k_transpose_w(const float* __restrict__ W, u16* __restrict__ WT, int K) {
  int idx = blockIdx.x * 256 + threadIdx.x;
  if (idx >= K * ED) return;
  int n = idx / K, k = idx % K;
  WT[idx] = f2b(W[(size_t)k * ED + n]);
}

// ---------------------------------------------------------------------------
// 2) GEMM: out[R,1024] = A[R,K](f32) x WT[1024,K](bf16) + bias
//    block = 256 thr = 8 waves (2 M x 4 N); tile 32 x 256; K-step 32
// ---------------------------------------------------------------------------
template <bool OUT_F32>
__global__ void __launch_bounds__(256)
k_gemm(const float* __restrict__ A, const u16* __restrict__ WT,
       const float* __restrict__ bias, float* __restrict__ outF,
       u16* __restrict__ outB, int K) {
  __shared__ u16 As[32][40];                   // 32 rows x 32 k, pad 8
  const int tid = threadIdx.x;
  const int wave = tid >> 5, lane = tid & 31;
  const int li = lane & 15, lg = lane >> 4;
  const int wm = wave >> 2, wn = wave & 3;
  const int rows0 = blockIdx.x * 32;
  const int cols0 = blockIdx.y * 256 + wn * 64;
  const int arow = wm * 16 + li;

  v8f acc[4];
#pragma unroll
  for (int j = 0; j < 4; ++j)
#pragma unroll
    for (int v = 0; v < 8; ++v) acc[j][v] = 0.f;

  for (int k0 = 0; k0 < K; k0 += 32) {
    __syncthreads();
    {
      int r = tid >> 3, c4 = (tid & 7) * 4;
      float4 av = *(const float4*)(A + (size_t)(rows0 + r) * K + k0 + c4);
      *(uint2*)&As[r][c4] = make_uint2(pack_bf16(av.x, av.y), pack_bf16(av.z, av.w));
    }
    __syncthreads();
    v16bf a = load_a_frag(&As[arow][0], lg);
#pragma unroll
    for (int j = 0; j < 4; ++j) {
      const u16* wrow = WT + (size_t)(cols0 + 16 * j + li) * K + k0;
      acc[j] = wmma_bf16(a, load_b_frag(wrow, lg), acc[j]);
    }
  }
#pragma unroll
  for (int j = 0; j < 4; ++j) {
    int col = cols0 + 16 * j + li;
    float bv = bias[col];
#pragma unroll
    for (int v = 0; v < 8; ++v) {
      int row = rows0 + wm * 16 + lg * 8 + v;
      float val = acc[j][v] + bv;
      if (OUT_F32) outF[(size_t)row * ED + col] = val;
      else         outB[(size_t)row * ED + col] = f2b(val);
    }
  }
}

// ---------------------------------------------------------------------------
// 3) full-row L2 normalize + split heads: in[row,1024] -> outH[b,h,n,64] bf16
// ---------------------------------------------------------------------------
template <bool IN_F32>
__global__ void __launch_bounds__(256)
k_rownorm(const float* __restrict__ inF, const u16* __restrict__ inB,
          u16* __restrict__ outH) {
  const int row = blockIdx.x;               // 0..4095
  const int b = row >> 11, n = row & 2047;
  const int tid = threadIdx.x;
  const int c0 = tid * 4;
  float vals[4];
  if (IN_F32) {
    float4 t = *(const float4*)(inF + (size_t)row * ED + c0);
    vals[0] = t.x; vals[1] = t.y; vals[2] = t.z; vals[3] = t.w;
  } else {
    ushort4 t = *(const ushort4*)(inB + (size_t)row * ED + c0);
    vals[0] = b2f(t.x); vals[1] = b2f(t.y); vals[2] = b2f(t.z); vals[3] = b2f(t.w);
  }
  float ss = vals[0]*vals[0] + vals[1]*vals[1] + vals[2]*vals[2] + vals[3]*vals[3];
  for (int m = 16; m >= 1; m >>= 1) ss += __shfl_xor(ss, m);
  __shared__ float sred[8];
  __shared__ float sinv;
  if ((tid & 31) == 0) sred[tid >> 5] = ss;
  __syncthreads();
  if (tid == 0) {
    float s = 0.f;
    for (int i = 0; i < 8; ++i) s += sred[i];
    sinv = 1.0f / fmaxf(sqrtf(s), 1e-12f);
  }
  __syncthreads();
  float inv = sinv;
  int h = c0 >> 6, d = c0 & 63;
  u16* dst = outH + (((size_t)(b * NH + h) * NSEQ) + n) * HD + d;
  ushort4 o = make_ushort4(f2b(vals[0] * inv), f2b(vals[1] * inv),
                           f2b(vals[2] * inv), f2b(vals[3] * inv));
  *(ushort4*)dst = o;
}

// ---------------------------------------------------------------------------
// 4) V transpose to [b,h,64,Nk] bf16 (LDS-tiled 32x32)
// ---------------------------------------------------------------------------
__global__ void __launch_bounds__(256)
k_vtrans(const u16* __restrict__ Vraw, u16* __restrict__ Vt) {
  __shared__ u16 tile[32][36];
  const int b = blockIdx.z;
  const int n0 = blockIdx.x * 32;
  const int c0 = blockIdx.y * 32;
  const int tid = threadIdx.x;
  {
    int r = tid >> 3, c = (tid & 7) * 4;
    ushort4 t = *(const ushort4*)(Vraw + ((size_t)(b * NSEQ + n0 + r)) * ED + c0 + c);
    tile[r][c] = t.x; tile[r][c+1] = t.y; tile[r][c+2] = t.z; tile[r][c+3] = t.w;
  }
  __syncthreads();
  {
    int dl = tid >> 3, nl = (tid & 7) * 4;
    int col = c0 + dl;
    int h = col >> 6, d = col & 63;
    u16* dst = Vt + (((size_t)(b * NH + h) * HD) + d) * NSEQ + n0 + nl;
    ushort4 o = make_ushort4(tile[nl][dl], tile[nl+1][dl], tile[nl+2][dl], tile[nl+3][dl]);
    *(ushort4*)dst = o;
  }
}

// ---------------------------------------------------------------------------
// 5) attention: grid (Nq/128, H, B); 8 waves x 16 q-rows; BK = 64
//    S^T = K x Q^T so the score D-layout IS the A-fragment layout of P for
//    the P·V WMMA (no transpose, no LDS round-trip). |score| <= 1/8 due to
//    l2-normalized Q,K => no online max; row sums reduce once at the end.
//    K/V tiles staged via GLOBAL_LOAD_ASYNC_TO_LDS (ASYNCcnt) when available.
// ---------------------------------------------------------------------------
__global__ void __launch_bounds__(256)
k_attn(const u16* __restrict__ Qh, const u16* __restrict__ Kh,
       const u16* __restrict__ Vt, u16* __restrict__ O) {
  __shared__ u16 Ks[64][72];        // 64 keys x 64 d
  __shared__ u16 Vs[64][72];        // 64 d   x 64 keys
  __shared__ float Lsum[8][16];     // per-wave row-sum broadcast
  const int tid = threadIdx.x;
  const int wave = tid >> 5, lane = tid & 31;
  const int li = lane & 15, lg = lane >> 4;
  const int b = blockIdx.z, h = blockIdx.y;
  const int q0 = blockIdx.x * 128 + wave * 16;
  const size_t bh = (size_t)(b * NH + h);
  const u16* Qb = Qh + bh * NSEQ * HD;
  const u16* Kb = Kh + bh * NSEQ * HD;
  const u16* Vb = Vt + bh * HD * NSEQ;

  // Q as B-fragments (lane = q column), once for the whole kernel
  v16bf qb[2];
#pragma unroll
  for (int t = 0; t < 2; ++t)
    qb[t] = load_b_frag(Qb + (size_t)(q0 + li) * HD + t * 32, lg);

  v8f o[4];
#pragma unroll
  for (int j = 0; j < 4; ++j)
#pragma unroll
    for (int v = 0; v < 8; ++v) o[j][v] = 0.f;
  float lsum = 0.f;                 // this lane's partial softmax denom (q = li)

  const int kr = tid >> 2, c8 = (tid & 3) * 8;  // 64 rows x 2 chunk-pairs
  for (int kb = 0; kb < NSEQ; kb += 64) {
    __syncthreads();
    async_b128(Kb + (size_t)(kb + kr) * HD + c8,      &Ks[kr][c8]);
    async_b128(Kb + (size_t)(kb + kr) * HD + c8 + 32, &Ks[kr][c8 + 32]);
    async_b128(Vb + (size_t)kr * NSEQ + kb + c8,      &Vs[kr][c8]);
    async_b128(Vb + (size_t)kr * NSEQ + kb + c8 + 32, &Vs[kr][c8 + 32]);
    async_wait0();
    __syncthreads();

#pragma unroll
    for (int half = 0; half < 2; ++half) {    // 32 keys per half
      // S^T tiles: rows = k, cols = q
      v8f st[2];
#pragma unroll
      for (int j = 0; j < 2; ++j) {
#pragma unroll
        for (int v = 0; v < 8; ++v) st[j][v] = 0.f;
#pragma unroll
        for (int t = 0; t < 2; ++t)
          st[j] = wmma_bf16(load_a_frag(&Ks[half * 32 + j * 16 + li][t * 32], lg),
                            qb[t], st[j]);
      }
      // raw-exp (arg in [-0.19,0.19]) + pack directly into P A-fragment
      float p0[8], p1[8];
#pragma unroll
      for (int v = 0; v < 8; ++v) {
        p0[v] = fast_exp2(st[0][v] * SCALE_LOG2E);
        p1[v] = fast_exp2(st[1][v] * SCALE_LOG2E);
        lsum += p0[v] + p1[v];
      }
      BF16Frag pf;
#pragma unroll
      for (int hh = 0; hh < 4; ++hh) {
        pf.w[hh]     = pack_bf16(p0[2 * hh], p0[2 * hh + 1]);
        pf.w[4 + hh] = pack_bf16(p1[2 * hh], p1[2 * hh + 1]);
      }
      // P·V: D rows = q, cols = d
#pragma unroll
      for (int j2 = 0; j2 < 4; ++j2)
        o[j2] = wmma_bf16(pf.v, load_b_frag(&Vs[j2 * 16 + li][half * 32], lg), o[j2]);
    }
  }

  // finish softmax denominator: combine the two lane groups, broadcast via LDS
  lsum += __shfl_xor(lsum, 16);
  if (lg == 0) Lsum[wave][li] = 1.0f / lsum;  // per-wave private, same-wave DS order
#pragma unroll
  for (int v = 0; v < 8; ++v) {
    float rinv = Lsum[wave][v + 8 * lg];      // row q = v + 8*lg
    int row = q0 + v + 8 * lg;
    u16* orow = O + ((size_t)(b * NSEQ) + row) * ED + h * HD;
#pragma unroll
    for (int j2 = 0; j2 < 4; ++j2)
      orow[j2 * 16 + li] = f2b(o[j2][v] * rinv);
  }
}

// ---------------------------------------------------------------------------
// 6) output projection + bias + residual(Qp) + LayerNorm; one WG owns 16 rows
// ---------------------------------------------------------------------------
__global__ void __launch_bounds__(256)
k_outln(const u16* __restrict__ Oin, const u16* __restrict__ WoT,
        const float* __restrict__ bo, const float* __restrict__ Qp,
        const float* __restrict__ gamma, const float* __restrict__ beta,
        float* __restrict__ out) {
  __shared__ u16 As[16][40];
  __shared__ float shS[16][8], shQ[16][8];
  __shared__ float shMu[16], shRs[16];
  const int tid = threadIdx.x;
  const int wave = tid >> 5, lane = tid & 31;
  const int li = lane & 15, lg = lane >> 4;
  const int r0 = blockIdx.x * 16;

  v8f acc[8];
#pragma unroll
  for (int j = 0; j < 8; ++j)
#pragma unroll
    for (int v = 0; v < 8; ++v) acc[j][v] = 0.f;

  for (int k0 = 0; k0 < ED; k0 += 32) {
    __syncthreads();
    if (tid < 64) {
      int r = tid >> 2, c8 = (tid & 3) * 8;
      async_b128(Oin + (size_t)(r0 + r) * ED + k0 + c8, &As[r][c8]);
    }
    async_wait0();
    __syncthreads();
    v16bf a = load_a_frag(&As[li][0], lg);
#pragma unroll
    for (int j = 0; j < 8; ++j) {
      int col = wave * 128 + j * 16 + li;
      acc[j] = wmma_bf16(a, load_b_frag(WoT + (size_t)col * ED + k0, lg), acc[j]);
    }
  }
  // bias + residual
#pragma unroll
  for (int j = 0; j < 8; ++j) {
    int col = wave * 128 + j * 16 + li;
    float bv = bo[col];
#pragma unroll
    for (int v = 0; v < 8; ++v) {
      int row = r0 + v + 8 * lg;
      acc[j][v] += bv + Qp[(size_t)row * ED + col];
    }
  }
  // LayerNorm reductions
#pragma unroll
  for (int v = 0; v < 8; ++v) {
    float s = 0.f, q = 0.f;
#pragma unroll
    for (int j = 0; j < 8; ++j) { float t = acc[j][v]; s += t; q += t * t; }
    s += __shfl_xor(s, 1); s += __shfl_xor(s, 2);
    s += __shfl_xor(s, 4); s += __shfl_xor(s, 8);
    q += __shfl_xor(q, 1); q += __shfl_xor(q, 2);
    q += __shfl_xor(q, 4); q += __shfl_xor(q, 8);
    if (li == 0) { shS[v + 8 * lg][wave] = s; shQ[v + 8 * lg][wave] = q; }
  }
  __syncthreads();
  if (tid < 16) {
    float s = 0.f, q = 0.f;
    for (int w = 0; w < 8; ++w) { s += shS[tid][w]; q += shQ[tid][w]; }
    float mu = s * (1.0f / 1024.0f);
    float var = q * (1.0f / 1024.0f) - mu * mu;
    shMu[tid] = mu;
    shRs[tid] = rsqrtf(var + 1e-5f);
  }
  __syncthreads();
#pragma unroll
  for (int v = 0; v < 8; ++v) {
    int rloc = v + 8 * lg;
    float mu = shMu[rloc], rs = shRs[rloc];
    int row = r0 + rloc;
#pragma unroll
    for (int j = 0; j < 8; ++j) {
      int col = wave * 128 + j * 16 + li;
      out[(size_t)row * ED + col] = (acc[j][v] - mu) * rs * gamma[col] + beta[col];
    }
  }
}

// ---------------------------------------------------------------------------
extern "C" void kernel_launch(void* const* d_in, const int* in_sizes, int n_in,
                              void* d_out, int out_size, void* d_ws, size_t ws_size,
                              hipStream_t stream) {
  (void)in_sizes; (void)n_in; (void)out_size; (void)ws_size;
  const float* query = (const float*)d_in[0];
  const float* key   = (const float*)d_in[1];
  const float* value = (const float*)d_in[2];
  const float* Wq    = (const float*)d_in[3];
  const float* bq    = (const float*)d_in[4];
  const float* Wk    = (const float*)d_in[5];
  const float* bk    = (const float*)d_in[6];
  const float* Wv    = (const float*)d_in[7];
  const float* bv    = (const float*)d_in[8];
  const float* Wo    = (const float*)d_in[9];
  const float* bo    = (const float*)d_in[10];
  const float* gamma = (const float*)d_in[11];
  const float* beta  = (const float*)d_in[12];
  float* out = (float*)d_out;

  char* ws = (char*)d_ws;
  size_t off = 0;
  auto alloc = [&](size_t bytes) -> void* {
    void* p = ws + off;
    off += (bytes + 255) & ~(size_t)255;
    return p;
  };
  const size_t R = (size_t)BATCH * NSEQ;        // 4096
  u16*   WqT  = (u16*)alloc((size_t)ED * QD * 2);
  u16*   WkT  = (u16*)alloc((size_t)ED * KVD * 2);
  u16*   WvT  = (u16*)alloc((size_t)ED * KVD * 2);
  u16*   WoT  = (u16*)alloc((size_t)ED * ED * 2);
  float* Qp   = (float*)alloc(R * ED * 4);
  u16*   Kraw = (u16*)alloc(R * ED * 2);
  u16*   Vraw = (u16*)alloc(R * ED * 2);
  u16*   Qhh  = (u16*)alloc(R * ED * 2);
  u16*   Khh  = (u16*)alloc(R * ED * 2);
  u16*   Vt   = (u16*)alloc(R * ED * 2);
  u16*   Oat  = (u16*)alloc(R * ED * 2);

  // 1) weights -> bf16 transposed
  k_transpose_w<<<(QD  * ED + 255) / 256, 256, 0, stream>>>(Wq, WqT, QD);
  k_transpose_w<<<(KVD * ED + 255) / 256, 256, 0, stream>>>(Wk, WkT, KVD);
  k_transpose_w<<<(KVD * ED + 255) / 256, 256, 0, stream>>>(Wv, WvT, KVD);
  k_transpose_w<<<(ED  * ED + 255) / 256, 256, 0, stream>>>(Wo, WoT, ED);

  // 2) projections
  dim3 gg(R / 32, ED / 256);
  k_gemm<true ><<<gg, 256, 0, stream>>>(query, WqT, bq, Qp, nullptr, QD);
  k_gemm<false><<<gg, 256, 0, stream>>>(key,   WkT, bk, nullptr, Kraw, KVD);
  k_gemm<false><<<gg, 256, 0, stream>>>(value, WvT, bv, nullptr, Vraw, KVD);

  // 3) l2 normalize (full embed dim) + head split
  k_rownorm<true ><<<(unsigned)R, 256, 0, stream>>>(Qp, nullptr, Qhh);
  k_rownorm<false><<<(unsigned)R, 256, 0, stream>>>(nullptr, Kraw, Khh);

  // 4) V -> [b,h,d,n]
  k_vtrans<<<dim3(NSEQ / 32, ED / 32, BATCH), 256, 0, stream>>>(Vraw, Vt);

  // 5) attention
  k_attn<<<dim3(NSEQ / 128, NH, BATCH), 256, 0, stream>>>(Qhh, Khh, Vt, Oat);

  // 6) output projection + residual + layernorm
  k_outln<<<R / 16, 256, 0, stream>>>(Oat, WoT, bo, Qp, gamma, beta, out);
}
